// NodeNNConvModel_45011257262124
// MI455X (gfx1250) — matrix operand order repaired and checked by reading, not compiled, not run on hardware
//
#include <hip/hip_runtime.h>

// ---------------------------------------------------------------------------
// MI455X / gfx1250 implementation of the NNConv GNN reference.
// Heavy math (edge-MLP GEMMs) mapped to v_wmma_f32_16x16x32_f16 (wave32).
// Packed W2 and gathered x[row] tiles staged into LDS via
// GLOBAL_LOAD_ASYNC_TO_LDS_B128 (ASYNCcnt), WMMA B fragments fed from LDS.
// ---------------------------------------------------------------------------

typedef __attribute__((ext_vector_type(16))) _Float16 v16h;
typedef __attribute__((ext_vector_type(8)))  _Float16 v8h;
typedef __attribute__((ext_vector_type(8)))  float    v8f;

#define NN 50000
#define NE 150000
#define HD 32
#define LEAKK 0.1f
#define BN_EPS 1e-5f

__device__ __forceinline__ float leaky_f(float v){ return v >= 0.f ? v : LEAKK * v; }

__device__ __forceinline__ v16h load_frag16(const _Float16* p0, const _Float16* p1){
  v8h a = *(const v8h*)p0;
  v8h b = *(const v8h*)p1;
  v16h r;
#pragma unroll
  for (int i = 0; i < 8; ++i){ r[i] = a[i]; r[8 + i] = b[i]; }
  return r;
}

// Issue one 16B global->LDS async copy (tracked by ASYNCcnt).
__device__ __forceinline__ void async_b128_to_lds(void* lds_ptr, const void* gptr){
  unsigned loff = (unsigned)(uintptr_t)lds_ptr;           // low 32 bits = LDS offset
  unsigned long long ga = (unsigned long long)(uintptr_t)gptr;
  asm volatile("global_load_async_to_lds_b128 %0, %1, off"
               :: "v"(loff), "v"(ga) : "memory");
}
__device__ __forceinline__ void wait_async0(){
  asm volatile("s_wait_asynccnt 0x0" ::: "memory");
}

// ---------------- BatchNorm statistics (wave32 shuffle reduce) -------------
template<int COLS>
__global__ void bn_stats_kernel(const float* __restrict__ src, int rows,
                                float* __restrict__ osum, float* __restrict__ osq){
  float ls[COLS], lq[COLS];
#pragma unroll
  for (int c = 0; c < COLS; ++c){ ls[c] = 0.f; lq[c] = 0.f; }
  for (int r = blockIdx.x * blockDim.x + threadIdx.x; r < rows; r += gridDim.x * blockDim.x){
#pragma unroll
    for (int c = 0; c < COLS; ++c){ float v = src[(size_t)r * COLS + c]; ls[c] += v; lq[c] += v * v; }
  }
  int lane = threadIdx.x & 31;
#pragma unroll
  for (int c = 0; c < COLS; ++c){
    float a = ls[c], b = lq[c];
    for (int off = 16; off > 0; off >>= 1){
      a += __shfl_down(a, off, 32);
      b += __shfl_down(b, off, 32);
    }
    if (lane == 0){ atomicAdd(&osum[c], a); atomicAdd(&osq[c], b); }
  }
}

// Normalize edges: write f32 [E,19] (for MetaLayer) and f16 [E,32] padded (WMMA A)
__global__ void bn_apply_edge_kernel(const float* __restrict__ e,
                                     const float* __restrict__ sum, const float* __restrict__ sq,
                                     const float* __restrict__ g, const float* __restrict__ b,
                                     float* __restrict__ ebn, _Float16* __restrict__ eh){
  int r = blockIdx.x * blockDim.x + threadIdx.x;
  if (r >= NE) return;
#pragma unroll
  for (int c = 0; c < 19; ++c){
    float m = sum[c] * (1.f / NE);
    float v = sq[c] * (1.f / NE) - m * m;
    float w = rsqrtf(v + BN_EPS);
    float val = (e[(size_t)r * 19 + c] - m) * w * g[c] + b[c];
    ebn[(size_t)r * 19 + c] = val;
    eh[(size_t)r * 32 + c] = (_Float16)val;
  }
#pragma unroll
  for (int c = 19; c < 32; ++c) eh[(size_t)r * 32 + c] = (_Float16)0.f;
}

// Normalize nodes into [N,32] buffer, cols 16..31 zero
__global__ void bn_apply_node_kernel(const float* __restrict__ x,
                                     const float* __restrict__ sum, const float* __restrict__ sq,
                                     const float* __restrict__ g, const float* __restrict__ b,
                                     float* __restrict__ xo){
  int r = blockIdx.x * blockDim.x + threadIdx.x;
  if (r >= NN) return;
#pragma unroll
  for (int c = 0; c < 16; ++c){
    float m = sum[c] * (1.f / NN);
    float v = sq[c] * (1.f / NN) - m * m;
    float w = rsqrtf(v + BN_EPS);
    xo[(size_t)r * 32 + c] = (x[(size_t)r * 16 + c] - m) * w * g[c] + b[c];
  }
#pragma unroll
  for (int c = 16; c < 32; ++c) xo[(size_t)r * 32 + c] = 0.f;
}

// ---------------- weight pre-packing into WMMA B-fragment order -----------
// B fragment order used here (contiguous k, same as the A loads): lane l: n=l&15, kb=(l<16)?0:16, b[i]=B[kb+i][n]
template<int DIN>
__global__ void pack_w1_kernel(const float* __restrict__ W1, _Float16* __restrict__ w1p){
  const int NCH = DIN / 16;
  int t = blockIdx.x * blockDim.x + threadIdx.x;
  if (t >= NCH * 512) return;
  int ch = t >> 9, lane = (t >> 4) & 31, i = t & 15;
  int n = lane & 15, kb = (lane < 16) ? 0 : 16, k = kb + i;
  int col = ch * 16 + n;
  float v = (k < 19) ? W1[(size_t)k * DIN + col] : 0.f;
  w1p[t] = (_Float16)v;
}

template<int DIN>
__global__ void pack_w2_kernel(const float* __restrict__ W2, _Float16* __restrict__ w2p){
  const int NCH = DIN * 2;
  int t = blockIdx.x * blockDim.x + threadIdx.x;
  if (t >= NCH * 512) return;
  int ch = t >> 9, lane = (t >> 4) & 31, i = t & 15;
  int n = lane & 15, kb = (lane < 16) ? 0 : 16, k = kb + i;
  int col = ch * 16 + n;
  float v = (k < DIN) ? W2[(size_t)k * (DIN * HD) + col] : 0.f;
  w2p[t] = (_Float16)v;
}

// ---------------- hidden = leaky(e_bn @ W1 + b1)  (WMMA) -------------------
template<int DIN>
__global__ void hidden_kernel(const _Float16* __restrict__ eh,
                              const _Float16* __restrict__ w1p,
                              const float* __restrict__ b1,
                              _Float16* __restrict__ hid){
  int lane = threadIdx.x & 31;
  int wave = threadIdx.x >> 5;
  int tile = blockIdx.x * (blockDim.x >> 5) + wave;
  if (tile >= NE / 16) return;
  int e0 = tile * 16;
  int m = lane & 15, s = (lane < 16) ? 0 : 8;
  int n = lane & 15, moff = (lane < 16) ? 0 : 8;
  const _Float16* ar = eh + (size_t)(e0 + m) * 32;
  v16h a = load_frag16(ar + 2 * s, ar + 2 * s + 8);
#pragma unroll
  for (int ch = 0; ch < DIN / 16; ++ch){
    const _Float16* bp = w1p + (size_t)(ch * 32 + lane) * 16;
    v16h b = load_frag16(bp, bp + 8);
    v8f c = {};
    c = __builtin_amdgcn_wmma_f32_16x16x32_f16(false, a, false, b, (short)0, c, false, false);
    int col = ch * 16 + n;
    float bias = b1[col];
#pragma unroll
    for (int r = 0; r < 8; ++r){
      float v = leaky_f(c[r] + bias);
      hid[(size_t)(e0 + r + moff) * 32 + col] = (_Float16)v;
    }
  }
  if (DIN == 16){
#pragma unroll
    for (int r = 0; r < 8; ++r)
      hid[(size_t)(e0 + r + moff) * 32 + 16 + n] = (_Float16)0.f;
  }
}

// -------- fused: w = leaky(hidden@W2+b2); msg = x[row]·w; scatter agg[col] --
// Packed W2 (whole layer) + gathered x[row] tiles staged in LDS via async
// global->LDS b128 copies; WMMA B fragments read from LDS (ds_load_b128).
template<int DIN>
__global__ void msg_kernel(const _Float16* __restrict__ hid,
                           const _Float16* __restrict__ w2p,
                           const float* __restrict__ b2,
                           const float* __restrict__ xin,
                           const int* __restrict__ ei,
                           float* __restrict__ agg){
  constexpr int NCH = DIN * 2;
  __shared__ _Float16 s_w2[NCH * 512];      // full packed W2: 32/64 KB
  __shared__ float    s_xr[8][16 * DIN];    // per-wave gathered x[row] tile
  __shared__ int      s_cl[8][16];
  int lane = threadIdx.x & 31;
  int wave = threadIdx.x >> 5;
  int tile = blockIdx.x * (blockDim.x >> 5) + wave;
  bool active = tile < NE / 16;
  int e0 = active ? tile * 16 : 0;

  // Stage packed W2 (block-wide): NCH*1024 bytes as 16B async transfers.
  for (int i = threadIdx.x; i < NCH * 64; i += blockDim.x)
    async_b128_to_lds(&s_w2[i * 8], w2p + (size_t)i * 8);

  if (active){
    // Gather x[row] tile: 16 edges x DIN floats, 16B per async transfer.
    constexpr int NV = 16 * DIN / 4;        // float4 transfers per tile
#pragma unroll
    for (int j = 0; j < NV / 32; ++j){
      int idx = j * 32 + lane;
      int er = idx / (DIN / 4);
      int q  = idx % (DIN / 4);
      int srcn = ei[e0 + er];
      async_b128_to_lds(&s_xr[wave][er * DIN + q * 4],
                        xin + (size_t)srcn * 32 + q * 4);
    }
    if (lane < 16) s_cl[wave][lane] = ei[NE + e0 + lane];  // col (target)
  }
  wait_async0();
  __syncthreads();
  if (!active) return;   // wave-uniform -> EXEC all ones for WMMA below

  int m = lane & 15, s = (lane < 16) ? 0 : 8;
  int n = lane & 15, moff = (lane < 16) ? 0 : 8;
  const _Float16* ar = hid + (size_t)(e0 + m) * 32;
  v16h a = load_frag16(ar + 2 * s, ar + 2 * s + 8);
  float accLo[8], accHi[8];
#pragma unroll
  for (int r = 0; r < 8; ++r){ accLo[r] = 0.f; accHi[r] = 0.f; }
  for (int ib = 0; ib < DIN; ++ib){
    float xv[8];
#pragma unroll
    for (int r = 0; r < 8; ++r) xv[r] = s_xr[wave][(r + moff) * DIN + ib];
    {
      const _Float16* bp = &s_w2[(size_t)((ib * 2) * 32 + lane) * 16];
      v16h b = load_frag16(bp, bp + 8);
      v8f c = {};
      c = __builtin_amdgcn_wmma_f32_16x16x32_f16(false, a, false, b, (short)0, c, false, false);
      float bias = b2[ib * 32 + n];
#pragma unroll
      for (int r = 0; r < 8; ++r){ float v = leaky_f(c[r] + bias); accLo[r] += xv[r] * v; }
    }
    {
      const _Float16* bp = &s_w2[(size_t)((ib * 2 + 1) * 32 + lane) * 16];
      v16h b = load_frag16(bp, bp + 8);
      v8f c = {};
      c = __builtin_amdgcn_wmma_f32_16x16x32_f16(false, a, false, b, (short)0, c, false, false);
      float bias = b2[ib * 32 + 16 + n];
#pragma unroll
      for (int r = 0; r < 8; ++r){ float v = leaky_f(c[r] + bias); accHi[r] += xv[r] * v; }
    }
  }
#pragma unroll
  for (int r = 0; r < 8; ++r){
    int cn = s_cl[wave][r + moff];
    atomicAdd(&agg[(size_t)cn * 32 + n],      accLo[r]);
    atomicAdd(&agg[(size_t)cn * 32 + 16 + n], accHi[r]);
  }
}

// ---------------- x = agg + x @ root + bias --------------------------------
template<int DIN>
__global__ void update_kernel(const float* __restrict__ xin, const float* __restrict__ agg,
                              const float* __restrict__ root, const float* __restrict__ bias,
                              float* __restrict__ xout){
  int v = blockIdx.x * blockDim.x + threadIdx.x;
  if (v >= NN) return;
  float acc[32];
#pragma unroll
  for (int o = 0; o < 32; ++o) acc[o] = agg[(size_t)v * 32 + o] + bias[o];
#pragma unroll
  for (int k = 0; k < DIN; ++k){
    float xv = xin[(size_t)v * 32 + k];
#pragma unroll
    for (int o = 0; o < 32; ++o) acc[o] += xv * root[k * 32 + o];
  }
#pragma unroll
  for (int o = 0; o < 32; ++o) xout[(size_t)v * 32 + o] = acc[o];
}

// ---------------- MetaLayer edge model + node-model-1 (fused, scatter) -----
__global__ void edge_meta_kernel(const float* __restrict__ x, const float* __restrict__ ebn,
                                 const int* __restrict__ ei,
                                 const float* __restrict__ emW1, const float* __restrict__ emb1,
                                 const float* __restrict__ emW2, const float* __restrict__ emb2,
                                 const float* __restrict__ n1W1, const float* __restrict__ n1b1,
                                 const float* __restrict__ n1W2, const float* __restrict__ n1b2,
                                 float* __restrict__ aggH, int* __restrict__ cnt){
  int e = blockIdx.x * blockDim.x + threadIdx.x;
  if (e >= NE) return;
  int rsrc = ei[e], ctgt = ei[NE + e];
  float xr[32];
#pragma unroll
  for (int o = 0; o < 32; ++o) xr[o] = x[(size_t)rsrc * 32 + o];
  float acc[32];
#pragma unroll
  for (int o = 0; o < 32; ++o) acc[o] = emb1[o];
#pragma unroll
  for (int k = 0; k < 32; ++k){
    float v = xr[k];
#pragma unroll
    for (int o = 0; o < 32; ++o) acc[o] += v * emW1[k * 32 + o];
  }
#pragma unroll
  for (int k = 0; k < 32; ++k){
    float v = x[(size_t)ctgt * 32 + k];
#pragma unroll
    for (int o = 0; o < 32; ++o) acc[o] += v * emW1[(32 + k) * 32 + o];
  }
#pragma unroll
  for (int k = 0; k < 19; ++k){
    float v = ebn[(size_t)e * 19 + k];
#pragma unroll
    for (int o = 0; o < 32; ++o) acc[o] += v * emW1[(64 + k) * 32 + o];
  }
  float t[32];
#pragma unroll
  for (int o = 0; o < 32; ++o) t[o] = leaky_f(acc[o]);
  float e2[32];
#pragma unroll
  for (int o = 0; o < 32; ++o) e2[o] = emb2[o];
#pragma unroll
  for (int k = 0; k < 32; ++k){
    float v = t[k];
#pragma unroll
    for (int o = 0; o < 32; ++o) e2[o] += v * emW2[k * 32 + o];
  }
  // node model 1: h = leaky([xr, e2] @ n1W1 + b1) @ n1W2 + b2
#pragma unroll
  for (int o = 0; o < 32; ++o) acc[o] = n1b1[o];
#pragma unroll
  for (int k = 0; k < 32; ++k){
    float v = xr[k];
#pragma unroll
    for (int o = 0; o < 32; ++o) acc[o] += v * n1W1[k * 32 + o];
  }
#pragma unroll
  for (int k = 0; k < 32; ++k){
    float v = e2[k];
#pragma unroll
    for (int o = 0; o < 32; ++o) acc[o] += v * n1W1[(32 + k) * 32 + o];
  }
#pragma unroll
  for (int o = 0; o < 32; ++o) t[o] = leaky_f(acc[o]);
  float h[32];
#pragma unroll
  for (int o = 0; o < 32; ++o) h[o] = n1b2[o];
#pragma unroll
  for (int k = 0; k < 32; ++k){
    float v = t[k];
#pragma unroll
    for (int o = 0; o < 32; ++o) h[o] += v * n1W2[k * 32 + o];
  }
#pragma unroll
  for (int o = 0; o < 32; ++o) atomicAdd(&aggH[(size_t)ctgt * 32 + o], h[o]);
  atomicAdd(&cnt[ctgt], 1);
}

// ---------------- node model 2 + log_softmax -------------------------------
__global__ void node_final_kernel(const float* __restrict__ x, const float* __restrict__ aggH,
                                  const int* __restrict__ cnt,
                                  const float* __restrict__ W1, const float* __restrict__ b1,
                                  const float* __restrict__ W2, const float* __restrict__ b2,
                                  float* __restrict__ out){
  int v = blockIdx.x * blockDim.x + threadIdx.x;
  if (v >= NN) return;
  float inv = 1.f / fmaxf((float)cnt[v], 1.f);
  float acc[32];
#pragma unroll
  for (int o = 0; o < 32; ++o) acc[o] = b1[o];
#pragma unroll
  for (int k = 0; k < 32; ++k){
    float xv = x[(size_t)v * 32 + k];
#pragma unroll
    for (int o = 0; o < 32; ++o) acc[o] += xv * W1[k * 32 + o];
  }
#pragma unroll
  for (int k = 0; k < 32; ++k){
    float av = aggH[(size_t)v * 32 + k] * inv;
#pragma unroll
    for (int o = 0; o < 32; ++o) acc[o] += av * W1[(32 + k) * 32 + o];
  }
  float z0 = b2[0], z1 = b2[1];
#pragma unroll
  for (int k = 0; k < 32; ++k){
    float u = leaky_f(acc[k]);
    z0 += u * W2[k * 2 + 0];
    z1 += u * W2[k * 2 + 1];
  }
  float mx = fmaxf(z0, z1);
  float l = mx + logf(expf(z0 - mx) + expf(z1 - mx));
  out[(size_t)v * 2 + 0] = z0 - l;
  out[(size_t)v * 2 + 1] = z1 - l;
}

// ---------------------------------------------------------------------------
extern "C" void kernel_launch(void* const* d_in, const int* in_sizes, int n_in,
                              void* d_out, int out_size, void* d_ws, size_t ws_size,
                              hipStream_t stream){
  (void)in_sizes; (void)n_in; (void)out_size; (void)ws_size;
  const float* x_in = (const float*)d_in[0];
  const int*   ei   = (const int*)d_in[1];
  const float* e_in = (const float*)d_in[2];
  const float* bng  = (const float*)d_in[4];
  const float* bnb  = (const float*)d_in[5];
  const float* beg  = (const float*)d_in[6];
  const float* beb  = (const float*)d_in[7];
  const float *W1s[3], *b1s[3], *W2s[3], *b2s[3], *roots[3], *biases[3];
  for (int l = 0; l < 3; ++l){
    int base = 8 + 6 * l;
    W1s[l]   = (const float*)d_in[base + 0];
    b1s[l]   = (const float*)d_in[base + 1];
    W2s[l]   = (const float*)d_in[base + 2];
    b2s[l]   = (const float*)d_in[base + 3];
    roots[l] = (const float*)d_in[base + 4];
    biases[l]= (const float*)d_in[base + 5];
  }
  const float* emW1 = (const float*)d_in[26];
  const float* emb1 = (const float*)d_in[27];
  const float* emW2 = (const float*)d_in[28];
  const float* emb2 = (const float*)d_in[29];
  const float* n1W1 = (const float*)d_in[30];
  const float* n1b1 = (const float*)d_in[31];
  const float* n1W2 = (const float*)d_in[32];
  const float* n1b2 = (const float*)d_in[33];
  const float* n2W1 = (const float*)d_in[34];
  const float* n2b1 = (const float*)d_in[35];
  const float* n2W2 = (const float*)d_in[36];
  const float* n2b2 = (const float*)d_in[37];

  char* ws = (char*)d_ws;
  size_t off = 0;
  auto take = [&](size_t bytes) -> void* {
    void* p = ws + off;
    off = (off + bytes + 255) & ~(size_t)255;
    return p;
  };
  float*    stats = (float*)take(256 * sizeof(float));
  float*    e_bn  = (float*)take((size_t)NE * 19 * 4);
  _Float16* e_h   = (_Float16*)take((size_t)NE * 32 * 2);
  float*    xA    = (float*)take((size_t)NN * 32 * 4);
  float*    xB    = (float*)take((size_t)NN * 32 * 4);
  _Float16* hid   = (_Float16*)take((size_t)NE * 32 * 2);
  float*    agg   = (float*)take((size_t)NN * 32 * 4);
  float*    aggH  = (float*)take((size_t)NN * 32 * 4);
  int*      cnt   = (int*)take((size_t)NN * 4);
  _Float16* w1p   = (_Float16*)take(2 * 512 * 2);
  _Float16* w2p   = (_Float16*)take(64 * 512 * 2);

  // BatchNorm (training-mode batch stats)
  hipMemsetAsync(stats, 0, 256 * sizeof(float), stream);
  bn_stats_kernel<19><<<120, 256, 0, stream>>>(e_in, NE, stats + 0,  stats + 19);
  bn_stats_kernel<16><<<120, 256, 0, stream>>>(x_in, NN, stats + 38, stats + 54);
  bn_apply_edge_kernel<<<(NE + 255) / 256, 256, 0, stream>>>(e_in, stats + 0, stats + 19, beg, beb, e_bn, e_h);
  bn_apply_node_kernel<<<(NN + 255) / 256, 256, 0, stream>>>(x_in, stats + 38, stats + 54, bng, bnb, xA);

  const int tiles = NE / 16;            // 9375
  const int tblocks = (tiles + 7) / 8;  // 8 waves/block

  // ---- layer 0 (din=16): xA -> xB ----
  pack_w1_kernel<16><<<2, 256, 0, stream>>>(W1s[0], w1p);
  pack_w2_kernel<16><<<(32 * 512 + 255) / 256, 256, 0, stream>>>(W2s[0], w2p);
  hidden_kernel<16><<<tblocks, 256, 0, stream>>>(e_h, w1p, b1s[0], hid);
  hipMemsetAsync(agg, 0, (size_t)NN * 32 * 4, stream);
  msg_kernel<16><<<tblocks, 256, 0, stream>>>(hid, w2p, b2s[0], xA, ei, agg);
  update_kernel<16><<<(NN + 255) / 256, 256, 0, stream>>>(xA, agg, roots[0], biases[0], xB);

  // ---- layer 1 (din=32): xB -> xA ----
  pack_w1_kernel<32><<<4, 256, 0, stream>>>(W1s[1], w1p);
  pack_w2_kernel<32><<<(64 * 512 + 255) / 256, 256, 0, stream>>>(W2s[1], w2p);
  hidden_kernel<32><<<tblocks, 256, 0, stream>>>(e_h, w1p, b1s[1], hid);
  hipMemsetAsync(agg, 0, (size_t)NN * 32 * 4, stream);
  msg_kernel<32><<<tblocks, 256, 0, stream>>>(hid, w2p, b2s[1], xB, ei, agg);
  update_kernel<32><<<(NN + 255) / 256, 256, 0, stream>>>(xB, agg, roots[1], biases[1], xA);

  // ---- layer 2 (din=32): xA -> xB ----
  pack_w1_kernel<32><<<4, 256, 0, stream>>>(W1s[2], w1p);
  pack_w2_kernel<32><<<(64 * 512 + 255) / 256, 256, 0, stream>>>(W2s[2], w2p);
  hidden_kernel<32><<<tblocks, 256, 0, stream>>>(e_h, w1p, b1s[2], hid);
  hipMemsetAsync(agg, 0, (size_t)NN * 32 * 4, stream);
  msg_kernel<32><<<tblocks, 256, 0, stream>>>(hid, w2p, b2s[2], xA, ei, agg);
  update_kernel<32><<<(NN + 255) / 256, 256, 0, stream>>>(xA, agg, roots[2], biases[2], xB);

  // ---- MetaLayer ----
  hipMemsetAsync(aggH, 0, (size_t)NN * 32 * 4, stream);
  hipMemsetAsync(cnt, 0, (size_t)NN * 4, stream);
  edge_meta_kernel<<<(NE + 255) / 256, 256, 0, stream>>>(xB, e_bn, ei,
      emW1, emb1, emW2, emb2, n1W1, n1b1, n1W2, n1b2, aggH, cnt);
  node_final_kernel<<<(NN + 255) / 256, 256, 0, stream>>>(xB, aggH, cnt,
      n2W1, n2b1, n2W2, n2b2, (float*)d_out);
}
